// RGCN_75574244540539
// MI455X (gfx1250) — matrix operand order repaired and checked by reading, not compiled, not run on hardware
//
#include <hip/hip_runtime.h>

typedef __attribute__((ext_vector_type(16))) _Float16 v16h;
typedef __attribute__((ext_vector_type(8)))  _Float16 v8h;
typedef __attribute__((ext_vector_type(8)))  float    v8f;

#define C        128
#define RNUM     7
#define TNUM     4
#define KB       (RNUM * C)     // 896
#define A_STRIDE 904            // 896 + 8 halfs pad -> 4-bank rotation per row
#define X_STRIDE 136            // 128 + 8

// ---------------------------------------------------------------------------
// Phase 1b: convert weights to f16 in [o][k] (column-contiguous for B frags)
//   Bh[o*KB + r*128 + c] = rel_w[r][o][c]
//   Rh[t*C*C + o*C + c]  = root_w[t][o][c]   (already the layout we want)
// ---------------------------------------------------------------------------
__global__ void rgcn_convert_w(const float* __restrict__ rel_w,
                               const float* __restrict__ root_w,
                               _Float16* __restrict__ Bh,
                               _Float16* __restrict__ Rh) {
  const int nB = KB * C;
  const int nR = TNUM * C * C;
  for (int i = blockIdx.x * blockDim.x + threadIdx.x; i < nB + nR;
       i += gridDim.x * blockDim.x) {
    if (i < nB) {
      int o = i / KB;
      int k = i - o * KB;
      int r = k >> 7;
      int c = k & 127;
      Bh[i] = (_Float16)rel_w[((size_t)r * C + o) * C + c];
    } else {
      int j = i - nB;
      Rh[j] = (_Float16)root_w[j];
    }
  }
}

// ---------------------------------------------------------------------------
// Phase 1c: pre-convert x to f16 (enables async DMA staging of X tiles and
// halves per-block X read traffic in the GEMM).
// ---------------------------------------------------------------------------
__global__ void rgcn_convert_x(const float* __restrict__ x,
                               _Float16* __restrict__ xh, int NC4) {
  int i = blockIdx.x * blockDim.x + threadIdx.x;
  if (i < NC4) {
    float4 v = ((const float4*)x)[i];
    _Float16* o = xh + (size_t)i * 4;
    o[0] = (_Float16)v.x;
    o[1] = (_Float16)v.y;
    o[2] = (_Float16)v.z;
    o[3] = (_Float16)v.w;
  }
}

// ---------------------------------------------------------------------------
// Phase 2: per-(node, edge_type) edge counts
// ---------------------------------------------------------------------------
__global__ void rgcn_count(const int* __restrict__ ei, const int* __restrict__ et,
                           unsigned* __restrict__ cnt, int E) {
  int e = blockIdx.x * blockDim.x + threadIdx.x;
  if (e < E) atomicAdd(&cnt[(size_t)ei[E + e] * RNUM + et[e]], 1u);
}

// ---------------------------------------------------------------------------
// Phase 3: scatter-add x[src] into sums[(dst, et)].  One wave32 per edge,
// each lane handles a float4 slice (x row is L2-resident: 51 MB < 192 MB L2).
// fp32 atomics keep the reduction exact.
// ---------------------------------------------------------------------------
__global__ void rgcn_scatter(const float* __restrict__ x, const int* __restrict__ ei,
                             const int* __restrict__ et, float* __restrict__ sums,
                             int E) {
  int lane = threadIdx.x & 31;
  int e = blockIdx.x * 8 + (threadIdx.x >> 5);
  if (e >= E) return;
  int src = ei[e];
  int dst = ei[E + e];
  int r   = et[e];
  const float4 v = *(const float4*)(x + (size_t)src * C + lane * 4);
  float* s = sums + ((size_t)dst * RNUM + r) * C + lane * 4;
  atomicAdd(s + 0, v.x);
  atomicAdd(s + 1, v.y);
  atomicAdd(s + 2, v.z);
  atomicAdd(s + 3, v.w);
}

// ---------------------------------------------------------------------------
// Phase 4: fused  out = (sums/cnt) @ Bh  +  masked per-type (x @ Rh_t) + bias
// 16-node tile per block; 8 waves each own a 16x16 output tile.
// X tile staged with the gfx1250 async Global->LDS DMA path (ASYNCcnt).
// ---------------------------------------------------------------------------
__global__ __launch_bounds__(256)
void rgcn_gemm(const float* __restrict__ sums, const unsigned* __restrict__ cnt,
               const _Float16* __restrict__ xh, const int* __restrict__ tnt,
               const _Float16* __restrict__ Bh, const _Float16* __restrict__ Rh,
               const float* __restrict__ root_b, float* __restrict__ out, int N) {
  __shared__ __align__(16) _Float16 Al[16 * A_STRIDE];
  __shared__ __align__(16) _Float16 Xl[16 * X_STRIDE];
  __shared__ int Tl[16];

  const int m0  = blockIdx.x * 16;
  const int tid = threadIdx.x;

  if (tid < 16) {
    int n = m0 + tid;
    Tl[tid] = (n < N) ? tnt[n] : 0;
  }

  // ---- X tile: one async b128 per lane moves the whole 4 KB tile ----
  {
    int row = tid >> 4;       // 0..15
    int seg = tid & 15;       // 16B (8-half) segment within the row
    int n   = m0 + row;
    _Float16* ldst = Xl + row * X_STRIDE + seg * 8;
    if (n < N) {
      unsigned lds_addr = (unsigned)(uintptr_t)ldst;   // generic low 32b == LDS offset
      unsigned long long ga =
          (unsigned long long)(uintptr_t)(xh + (size_t)n * C + seg * 8);
      asm volatile("global_load_async_to_lds_b128 %0, %1, off"
                   :: "v"(lds_addr), "v"(ga)
                   : "memory");
    } else {
#pragma unroll
      for (int j = 0; j < 8; ++j) ldst[j] = (_Float16)0.f;
    }
  }

  // ---- A tile: means (sums * 1/cnt) converted to f16 (needs VALU anyway) ----
  for (int i = tid; i < 16 * (KB / 4); i += 256) {
    int row = i / (KB / 4);
    int k   = (i - row * (KB / 4)) * 4;
    int n   = m0 + row;
    float4 v = make_float4(0.f, 0.f, 0.f, 0.f);
    float  s = 0.f;
    if (n < N) {
      v = *(const float4*)(sums + (size_t)n * KB + k);
      unsigned c = cnt[(size_t)n * RNUM + (k >> 7)];
      s = 1.0f / (float)(c > 1u ? c : 1u);
    }
    _Float16* a = Al + row * A_STRIDE + k;
    a[0] = (_Float16)(v.x * s);
    a[1] = (_Float16)(v.y * s);
    a[2] = (_Float16)(v.z * s);
    a[3] = (_Float16)(v.w * s);
  }

  asm volatile("s_wait_asynccnt 0x0" ::: "memory");
  __syncthreads();

  const int lane = tid & 31;
  const int hi   = lane >> 4;   // which K-half of the fragment this lane holds
  const int ln   = lane & 15;   // M (for A) / N (for B) index
  const int oc0  = (tid >> 5) * 16;

  // ---- relation GEMM: K = 896, 28 WMMAs ----
  const _Float16* Arow = Al + ln * A_STRIDE;
  v8f acc = {0.f, 0.f, 0.f, 0.f, 0.f, 0.f, 0.f, 0.f};
#pragma unroll 4
  for (int kb = 0; kb < KB; kb += 32) {
    v8h a0 = *(const v8h*)(Arow + kb + hi * 16);
    v8h a1 = *(const v8h*)(Arow + kb + hi * 16 + 8);
    v16h av = __builtin_shufflevector(a0, a1, 0, 1, 2, 3, 4, 5, 6, 7,
                                      8, 9, 10, 11, 12, 13, 14, 15);
    const _Float16* bp = Bh + (size_t)(oc0 + ln) * KB + kb + hi * 16;
    v8h b0 = *(const v8h*)bp;
    v8h b1 = *(const v8h*)(bp + 8);
    v16h bv = __builtin_shufflevector(b0, b1, 0, 1, 2, 3, 4, 5, 6, 7,
                                      8, 9, 10, 11, 12, 13, 14, 15);
    acc = __builtin_amdgcn_wmma_f32_16x16x32_f16(false, av, false, bv,
                                                 (short)0, acc, false, false);
  }

  // ---- root linear: 4 node types, masked per row ----
  const _Float16* Xrow = Xl + ln * X_STRIDE;
#pragma unroll
  for (int t = 0; t < TNUM; ++t) {
    v8f rc = {0.f, 0.f, 0.f, 0.f, 0.f, 0.f, 0.f, 0.f};
#pragma unroll
    for (int kb = 0; kb < C; kb += 32) {
      v8h a0 = *(const v8h*)(Xrow + kb + hi * 16);
      v8h a1 = *(const v8h*)(Xrow + kb + hi * 16 + 8);
      v16h av = __builtin_shufflevector(a0, a1, 0, 1, 2, 3, 4, 5, 6, 7,
                                        8, 9, 10, 11, 12, 13, 14, 15);
      const _Float16* bp = Rh + ((size_t)t * C + oc0 + ln) * C + kb + hi * 16;
      v8h b0 = *(const v8h*)bp;
      v8h b1 = *(const v8h*)(bp + 8);
      v16h bv = __builtin_shufflevector(b0, b1, 0, 1, 2, 3, 4, 5, 6, 7,
                                        8, 9, 10, 11, 12, 13, 14, 15);
      rc = __builtin_amdgcn_wmma_f32_16x16x32_f16(false, av, false, bv,
                                                  (short)0, rc, false, false);
    }
#pragma unroll
    for (int i = 0; i < 8; ++i) {
      int row = i + hi * 8;
      if (Tl[row] == t) acc[i] += rc[i];
    }
  }

  // ---- epilogue: bias + store ----
#pragma unroll
  for (int i = 0; i < 8; ++i) {
    int row = i + hi * 8;
    int n   = m0 + row;
    if (n < N) {
      int col = oc0 + ln;
      out[(size_t)n * C + col] = acc[i] + root_b[(size_t)Tl[row] * C + col];
    }
  }
}

// ---------------------------------------------------------------------------
extern "C" void kernel_launch(void* const* d_in, const int* in_sizes, int n_in,
                              void* d_out, int out_size, void* d_ws, size_t ws_size,
                              hipStream_t stream) {
  const float* x      = (const float*)d_in[0];
  const int*   ei     = (const int*)d_in[1];
  const int*   et     = (const int*)d_in[2];
  const int*   tnt    = (const int*)d_in[3];
  const float* rel_w  = (const float*)d_in[4];
  const float* root_w = (const float*)d_in[5];
  const float* root_b = (const float*)d_in[6];
  float*       out    = (float*)d_out;

  const int N = in_sizes[0] / C;   // 100000
  const int E = in_sizes[1] / 2;   // 1600000

  // workspace layout (256B aligned slabs)
  size_t off = 0;
  auto take = [&](size_t bytes) -> void* {
    void* p = (char*)d_ws + off;
    off = (off + bytes + 255) & ~(size_t)255;
    return p;
  };
  float*     sums = (float*)take((size_t)N * RNUM * C * sizeof(float)); // 358.4 MB
  unsigned*  cnt  = (unsigned*)take((size_t)N * RNUM * sizeof(unsigned));
  _Float16*  Bh   = (_Float16*)take((size_t)KB * C * sizeof(_Float16));
  _Float16*  Rh   = (_Float16*)take((size_t)TNUM * C * C * sizeof(_Float16));
  _Float16*  xh   = (_Float16*)take((size_t)N * C * sizeof(_Float16));  // 25.6 MB

  hipMemsetAsync(sums, 0, (size_t)N * RNUM * C * sizeof(float), stream);
  hipMemsetAsync(cnt, 0, (size_t)N * RNUM * sizeof(unsigned), stream);

  const int convElems = KB * C + TNUM * C * C;
  rgcn_convert_w<<<(convElems + 255) / 256, 256, 0, stream>>>(rel_w, root_w, Bh, Rh);
  const int NC4 = N * C / 4;
  rgcn_convert_x<<<(NC4 + 255) / 256, 256, 0, stream>>>(x, xh, NC4);
  rgcn_count<<<(E + 255) / 256, 256, 0, stream>>>(ei, et, cnt, E);
  rgcn_scatter<<<(E + 7) / 8, 256, 0, stream>>>(x, ei, et, sums, E);
  rgcn_gemm<<<(N + 15) / 16, 256, 0, stream>>>(sums, cnt, xh, tnt, Bh, Rh,
                                               root_b, out, N);
}
